// MFFN_36490042147633
// MI455X (gfx1250) — matrix operand order repaired and checked by reading, not compiled, not run on hardware
//
#include <hip/hip_runtime.h>
#include <hip/hip_bf16.h>
#include <math.h>

// ---------------------------------------------------------------------------
// MI455X (gfx1250) fused MFFN:
//   GEMM1 (64->256) -> dw3x3 / dw5x5 -> GELU gate -> GEMM2 (128->64)
//   -> per-8x8-patch spectral filter (== per-channel 8x8 circular conv).
// GEMMs on v_wmma_f32_16x16x32_bf16 (f32 accumulate). B (weight) fragments
// register-resident per wave; A fragments reused across 2 N-tiles; packed
// b64 LDS writeback of accumulators.
// ---------------------------------------------------------------------------

typedef __attribute__((ext_vector_type(16))) __bf16 v16bf;
typedef __attribute__((ext_vector_type(8)))  float  v8f;

#define NTHREADS 256

// LDS overlay offsets (bytes). Total = 160,384 B.
#define OFF_AX     0        // sA_x  : 144 px * 68 bf16  = 19,584 B   (phase 0/1)
#define OFF_OUT    0        // sOut  : 64 ch * 66 f32    = 16,896 B   (phase 3/4)
#define OFF_BWIN   19584    // sB_win: 256 * 68 bf16     = 34,816 B   (phase 0/1)
#define OFF_G      19584    // sG    : 64 px * 136 bf16  = 17,408 B   (phase 2/3)
#define OFF_K      36992    // sK    : 64*64 f32         = 16,384 B   (phase 2..4)
#define OFF_H      54400    // sH    : 256 ch * 148 bf16 = 75,776 B
#define OFF_BWOUT  130176   // sB_wout: 64 * 132 bf16    = 16,896 B
#define OFF_DW3    147072   // 192*9 f32 = 6,912 B
#define OFF_DW5    153984   // 64*25 f32 = 6,400 B
#define SMEM_BYTES 160384

union Frag { v16bf v; unsigned u[8]; };

// A fragment (16x32 bf16, M=rows): lane holds row (lane&15); lanes 0-15 carry
// K=0..15 (2 per VGPR), lanes 16-31 carry K=16..31.
__device__ __forceinline__ void load_frag_a(const __bf16* p, int rowoff,
                                            int kbase, int koff, Frag& f) {
#pragma unroll
    for (int r2 = 0; r2 < 8; ++r2) {
        int ka = kbase + 2 * r2 + koff;
        f.u[r2] = *(const unsigned*)(p + rowoff + ka);
    }
}

// B fragment (32x16 bf16, N=cols): lane holds col (lane&15); lanes 0-15 carry
// K=0..15 (2 per VGPR), lanes 16-31 carry K=16..31.
__device__ __forceinline__ void load_frag_b(const __bf16* p, int rowoff,
                                            int kbase, int kh, Frag& f) {
#pragma unroll
    for (int r2 = 0; r2 < 8; ++r2)
        f.u[r2] = *(const unsigned*)(p + rowoff + kbase + kh + 2 * r2);
}

// Pack 8 f32 accumulator lanes (8 consecutive pixels) into bf16 pairs and
// store as two b64 LDS writes (channel stride is 8-byte aligned).
__device__ __forceinline__ void store_acc_bf16(__bf16* dst, const v8f& acc) {
    union { __bf16 h[8]; uint2 d[2]; } pk;
#pragma unroll
    for (int r2 = 0; r2 < 8; ++r2) pk.h[r2] = (__bf16)acc[r2];
    *(uint2*)(dst + 0) = pk.d[0];
    *(uint2*)(dst + 4) = pk.d[1];
}

// ---------------------------------------------------------------------------
// Kernel 0: spectral kernel build.  k_c = irfft2(w_spec[c]) over 8x8,
// via Hermitian extension of the 8x5 half-spectrum. 64 blocks x 64 threads.
// ---------------------------------------------------------------------------
__global__ void mffn_build_spectral(const float* __restrict__ w_spec,
                                    float* __restrict__ kout) {
    int c = blockIdx.x;           // channel 0..63
    int t = threadIdx.x;          // u*8+v
    int u = t >> 3, v = t & 7;
    float acc = 0.f;
    for (int a = 0; a < 8; ++a) {
        for (int b = 0; b < 8; ++b) {
            float Fe = (b <= 4) ? w_spec[c * 40 + a * 5 + b]
                                : w_spec[c * 40 + ((8 - a) & 7) * 5 + (8 - b)];
            int ph = (a * u + b * v) & 7;
            acc += Fe * __cosf(0.78539816339744831f * (float)ph); // 2*pi/8
        }
    }
    kout[c * 64 + t] = acc * (1.0f / 64.0f);
}

// ---------------------------------------------------------------------------
// Main fused kernel: one workgroup (8 waves) per 8x8 output patch.
// ---------------------------------------------------------------------------
__global__ __launch_bounds__(NTHREADS, 1)
void mffn_fused(const float* __restrict__ x,      // (4,64,256,256)
                const float* __restrict__ w_in,   // (256,64)
                const float* __restrict__ dw3,    // (192,1,3,3)
                const float* __restrict__ dw5,    // (64,1,5,5)
                const float* __restrict__ w_out,  // (64,128)
                const float* __restrict__ kspec,  // (64,8,8) from kernel 0
                float* __restrict__ out)          // (4,64,256,256)
{
    __shared__ __attribute__((aligned(16))) unsigned char smem[SMEM_BYTES];
    __bf16* sA_x   = (__bf16*)(smem + OFF_AX);
    __bf16* sB_win = (__bf16*)(smem + OFF_BWIN);
    __bf16* sH     = (__bf16*)(smem + OFF_H);
    __bf16* sB_wout= (__bf16*)(smem + OFF_BWOUT);
    __bf16* sG     = (__bf16*)(smem + OFF_G);
    float*  sOut   = (float*) (smem + OFF_OUT);
    float*  sK     = (float*) (smem + OFF_K);
    float*  sDW3   = (float*) (smem + OFF_DW3);
    float*  sDW5   = (float*) (smem + OFF_DW5);

    const int tid  = threadIdx.x;
    const int lane = tid & 31;
    const int wave = tid >> 5;

    const int id  = blockIdx.x;        // 4096 patches
    const int bat = id >> 10;          // batch 0..3
    const int py  = (id >> 5) & 31;    // patch row 0..31
    const int pxc = id & 31;           // patch col 0..31
    const int row0 = py * 8 - 2;       // tile origin (halo 2)
    const int col0 = pxc * 8 - 2;

    const int mr   = (lane & 15);
    const int koff = (lane < 16) ? 0 : 16;  // A K-offset
    const int kh   = (lane < 16) ? 0 : 16;  // B K-half
    const int dsel = (lane < 16) ? 0 : 8;   // D M-offset

    // ---------------- Phase 0: stage x tile + weights (f32 -> bf16) --------
    for (int idx = tid; idx < 144 * 64; idx += NTHREADS) {
        int ch = idx / 144, px = idx - ch * 144;      // px-major for coalescing
        int r = row0 + px / 12, cc = col0 + px % 12;
        float v = 0.f;
        if ((unsigned)r < 256u && (unsigned)cc < 256u)
            v = x[(((bat * 64 + ch) * 256 + r) * 256) + cc];
        sA_x[px * 68 + ch] = (__bf16)v;
    }
    for (int idx = tid; idx < 256 * 64; idx += NTHREADS) {
        int o = idx >> 6, c = idx & 63;
        sB_win[o * 68 + c] = (__bf16)w_in[idx];
    }
    for (int idx = tid; idx < 64 * 128; idx += NTHREADS) {
        int o = idx >> 7, c = idx & 127;
        sB_wout[o * 132 + c] = (__bf16)w_out[idx];
    }
    for (int idx = tid; idx < 192 * 9;  idx += NTHREADS) sDW3[idx] = dw3[idx];
    for (int idx = tid; idx < 64 * 25;  idx += NTHREADS) sDW5[idx] = dw5[idx];
    __syncthreads();

    // ---------------- Phase 1: GEMM1  h[144px,256ch] = x @ w_in^T ----------
    // Each wave owns 2 N-tiles (16 ch each); B fragments register-resident.
    // 9 M-tiles streamed; A fragments reused across the 2 N-tiles.
    {
        const int nt0 = wave * 2;               // N-tiles: nt0, nt0+1
        Frag b00, b01, b10, b11;
        load_frag_b(sB_win, (nt0 * 16 + mr) * 68,       0,  kh, b00);
        load_frag_b(sB_win, (nt0 * 16 + mr) * 68,       32, kh, b01);
        load_frag_b(sB_win, ((nt0 + 1) * 16 + mr) * 68, 0,  kh, b10);
        load_frag_b(sB_win, ((nt0 + 1) * 16 + mr) * 68, 32, kh, b11);
        const int ch0 = (nt0 * 16 + mr) * 148;
        const int ch1 = ((nt0 + 1) * 16 + mr) * 148;
        for (int mt = 0; mt < 9; ++mt) {
            Frag a0, a1;
            load_frag_a(sA_x, (mt * 16 + mr) * 68, 0,  koff, a0);
            load_frag_a(sA_x, (mt * 16 + mr) * 68, 32, koff, a1);
            v8f acc0 = {0.f, 0.f, 0.f, 0.f, 0.f, 0.f, 0.f, 0.f};
            v8f acc1 = {0.f, 0.f, 0.f, 0.f, 0.f, 0.f, 0.f, 0.f};
            acc0 = __builtin_amdgcn_wmma_f32_16x16x32_bf16(
                       false, a0.v, false, b00.v, (short)0, acc0, false, false);
            acc0 = __builtin_amdgcn_wmma_f32_16x16x32_bf16(
                       false, a1.v, false, b01.v, (short)0, acc0, false, false);
            acc1 = __builtin_amdgcn_wmma_f32_16x16x32_bf16(
                       false, a0.v, false, b10.v, (short)0, acc1, false, false);
            acc1 = __builtin_amdgcn_wmma_f32_16x16x32_bf16(
                       false, a1.v, false, b11.v, (short)0, acc1, false, false);
            int pxb = mt * 16 + dsel;               // D: M=r2(+8), N=lane%16
            store_acc_bf16(sH + ch0 + pxb, acc0);
            store_acc_bf16(sH + ch1 + pxb, acc1);
        }
    }
    __syncthreads();

    // ---------------- Phase 2: depthwise + GELU gate -> g[64px,128ch] ------
    // (also stage spectral kernels into freed sB_win space)
    for (int idx = tid; idx < 64 * 64; idx += NTHREADS) sK[idx] = kspec[idx];

    for (int oidx = tid; oidx < 128 * 64; oidx += NTHREADS) {
        int ch = oidx >> 6, px = oidx & 63;
        int u = px >> 3, v = px & 7;
        int rb = 2 + u, cb = 2 + v;                 // interior offset in 12x12
        float gate = 0.f;                           // gate = dw3(h[ch])
#pragma unroll
        for (int ki = 0; ki < 3; ++ki)
#pragma unroll
            for (int kj = 0; kj < 3; ++kj)
                gate += (float)sH[ch * 148 + (rb - 1 + ki) * 12 + (cb - 1 + kj)]
                        * sDW3[ch * 9 + ki * 3 + kj];
        float mul;
        if (ch < 64) {                              // x1_2 = dw3(h[128+ch])
            int c2 = 128 + ch;
            float m = 0.f;
#pragma unroll
            for (int ki = 0; ki < 3; ++ki)
#pragma unroll
                for (int kj = 0; kj < 3; ++kj)
                    m += (float)sH[c2 * 148 + (rb - 1 + ki) * 12 + (cb - 1 + kj)]
                         * sDW3[c2 * 9 + ki * 3 + kj];
            mul = m;
        } else {                                    // y2 = dw5(h[192+ch-64])
            int c5 = ch - 64;
            int c2 = 192 + c5;
            float m = 0.f;
#pragma unroll
            for (int ki = 0; ki < 5; ++ki)
#pragma unroll
                for (int kj = 0; kj < 5; ++kj)
                    m += (float)sH[c2 * 148 + (rb - 2 + ki) * 12 + (cb - 2 + kj)]
                         * sDW5[c5 * 25 + ki * 5 + kj];
            mul = m;
        }
        float ge = 0.5f * gate * (1.0f + erff(gate * 0.70710678118654752f));
        sG[px * 136 + ch] = (__bf16)(ge * mul);
    }
    __syncthreads();

    // ---------------- Phase 3: GEMM2  out[64px,64ch] = g @ w_out^T ---------
    // wave -> M-tile (wave>>1) and 2 N-tiles ((wave&1)*2 ..). A fragments
    // (K=128: 4 chunks) register-resident, reused across both N-tiles.
    {
        const int mt  = wave >> 1;
        const int ntb = (wave & 1) * 2;
        Frag a[4];
#pragma unroll
        for (int kc = 0; kc < 4; ++kc)
            load_frag_a(sG, (mt * 16 + mr) * 136, kc * 32, koff, a[kc]);
        const int pxb = mt * 16 + dsel;
#pragma unroll
        for (int nti = 0; nti < 2; ++nti) {
            int nt = ntb + nti;
            v8f acc = {0.f, 0.f, 0.f, 0.f, 0.f, 0.f, 0.f, 0.f};
#pragma unroll
            for (int kc = 0; kc < 4; ++kc) {
                Frag bf;
                load_frag_b(sB_wout, (nt * 16 + mr) * 132, kc * 32, kh, bf);
                acc = __builtin_amdgcn_wmma_f32_16x16x32_bf16(
                          false, a[kc].v, false, bf.v, (short)0, acc, false, false);
            }
            float* d = sOut + (nt * 16 + mr) * 66 + pxb;  // f32, [ch][px]
#pragma unroll
            for (int r2 = 0; r2 < 4; ++r2) {        // 8-byte packed stores
                float2 p2; p2.x = acc[2 * r2]; p2.y = acc[2 * r2 + 1];
                *(float2*)(d + 2 * r2) = p2;
            }
        }
    }
    __syncthreads();

    // ---------------- Phase 4: 8x8 circular conv (spectral) + store --------
    for (int oidx = tid; oidx < 64 * 64; oidx += NTHREADS) {
        int c = oidx >> 6, px = oidx & 63;
        int u = px >> 3, v = px & 7;
        float acc = 0.f;
#pragma unroll
        for (int p = 0; p < 8; ++p)
#pragma unroll
            for (int q = 0; q < 8; ++q)
                acc += sOut[c * 66 + p * 8 + q]
                       * sK[c * 64 + ((u - p) & 7) * 8 + ((v - q) & 7)];
        out[(((bat * 64 + c) * 256 + (py * 8 + u)) * 256) + (pxc * 8 + v)] = acc;
    }
}

// ---------------------------------------------------------------------------
extern "C" void kernel_launch(void* const* d_in, const int* in_sizes, int n_in,
                              void* d_out, int out_size, void* d_ws, size_t ws_size,
                              hipStream_t stream) {
    const float* x      = (const float*)d_in[0];
    const float* w_in   = (const float*)d_in[1];
    const float* dw3    = (const float*)d_in[2];
    const float* dw5    = (const float*)d_in[3];
    const float* w_out  = (const float*)d_in[4];
    const float* w_spec = (const float*)d_in[5];
    float* out   = (float*)d_out;
    float* kspec = (float*)d_ws;   // 64*64 f32 = 16 KB

    mffn_build_spectral<<<64, 64, 0, stream>>>(w_spec, kspec);
    mffn_fused<<<4096, NTHREADS, 0, stream>>>(x, w_in, dw3, dw5, w_out, kspec, out);
}
